// OCAB_80831284510938
// MI455X (gfx1250) — matrix operand order repaired and checked by reading, not compiled, not run on hardware
//
#include <hip/hip_runtime.h>
#include <hip/hip_bf16.h>
#include <math.h>

// ---------------- types ----------------
typedef __attribute__((ext_vector_type(16))) __bf16 v16bf;
typedef __attribute__((ext_vector_type(8)))  __bf16 v8bf;
typedef __attribute__((ext_vector_type(8)))  float  v8f;
typedef __attribute__((ext_vector_type(4)))  int    v4i;

#define AS1 __attribute__((address_space(1)))
#define AS3 __attribute__((address_space(3)))

#if __has_builtin(__builtin_amdgcn_global_load_async_to_lds_b128) && \
    __has_builtin(__builtin_amdgcn_s_wait_asynccnt)
#define USE_ASYNC_LDS 1
#endif

// ---------------- problem constants ----------------
constexpr int   Bq     = 4;
constexpr int   Hdim   = 192;
constexpr int   Wdim   = 192;
constexpr int   NHd    = 6;
constexpr int   WSz    = 8;
constexpr int   OWSz   = 12;
constexpr int   PADz   = 2;
constexpr int   PH     = Hdim + 2 * PADz;       // 196
constexpr int   NWH    = Hdim / WSz;            // 24
constexpr int   NWIN   = Bq * NWH * NWH;        // 2304
constexpr long  NTOK   = (long)Bq * Hdim * Wdim;  // 147456
constexpr float SCALE_Q = 0.25f;                // HD^-0.5
constexpr float LN_EPS  = 1e-5f;

// ---------------- workspace layout (bytes) ----------------
constexpr size_t SZ_XN   = (size_t)NTOK * 96 * 2;        // bf16 N x 96
constexpr size_t SZ_KPAD = (size_t)Bq * PH * PH * 96 * 2;
constexpr size_t SZ_A1   = (size_t)NTOK * 192 * 2;       // holds xn -> attn_out -> h1 -> h3
constexpr size_t SZ_A2   = (size_t)NTOK * 384 * 2;       // holds q+kpad+vpad -> h2
constexpr size_t O_A1 = 0;
constexpr size_t O_A2 = O_A1 + SZ_A1;
constexpr size_t O_X2 = O_A2 + SZ_A2;                    // fp32 N x 96
constexpr size_t O_YN = O_X2 + (size_t)NTOK * 96 * 4;    // bf16 N x 96
constexpr size_t O_WT = O_YN + SZ_XN;                    // transposed bf16 weights
constexpr size_t O_Q  = O_A2;
constexpr size_t O_KP = O_A2 + SZ_XN;
constexpr size_t O_VP = O_KP + SZ_KPAD;
constexpr size_t O_WQKV  = O_WT;                 // 288x96  bf16 = 55296 B
constexpr size_t O_WPROJ = O_WQKV  + 55296;      // 96x96   = 18432
constexpr size_t O_WFC1  = O_WPROJ + 18432;      // 192x96  = 36864
constexpr size_t O_WFC2  = O_WFC1  + 36864;      // 384x192 = 147456
constexpr size_t O_WFC3  = O_WFC2  + 147456;     // 192x384 = 147456
constexpr size_t O_WFC4  = O_WFC3  + 147456;     // 96x192  = 36864

// ---------------- fragment helpers ----------------
// A frag (16-bit 16x32): p points at row + k0 + (lane/16)*16
static __device__ __forceinline__ v16bf ldA(const __bf16* p) {
  v8bf lo = *(const v8bf*)p;
  v8bf hi = *(const v8bf*)(p + 8);
  v16bf r;
#pragma unroll
  for (int i = 0; i < 8; ++i) { r[i] = lo[i]; r[i + 8] = hi[i]; }
  return r;
}
// A frag with K = 16 zero-padded to 32: only k = (lane/16)*8 + 0..7 valid
static __device__ __forceinline__ v16bf ldA_k16(const __bf16* p) {
  v8bf lo = *(const v8bf*)p;
  v16bf r;
#pragma unroll
  for (int i = 0; i < 8; ++i) { r[i] = lo[i]; r[i + 8] = (__bf16)0.0f; }
  return r;
}
// B frag (16-bit 32x16): lane holds 16 contiguous K at p = col_row + k0 + (lane/16)*16
static __device__ __forceinline__ v16bf ldB(const __bf16* p) {
  v8bf lo = *(const v8bf*)p;
  v8bf hi = *(const v8bf*)(p + 8);
  v16bf r;
#pragma unroll
  for (int i = 0; i < 8; ++i) { r[i] = lo[i]; r[i + 8] = hi[i]; }
  return r;
}
static __device__ __forceinline__ v16bf zero16() {
  v16bf r;
#pragma unroll
  for (int i = 0; i < 16; ++i) r[i] = (__bf16)0.0f;
  return r;
}
static __device__ __forceinline__ float gelu_exact(float v) {
  return 0.5f * v * (1.0f + erff(v * 0.70710678118654752f));
}

// ---------------- weight convert + transpose: W[K][M] f32 -> Wt[M][K] bf16 ----------------
__global__ void ocab_cvtw(const float* __restrict__ w, __bf16* __restrict__ wt, int K, int M) {
  int idx = blockIdx.x * 256 + threadIdx.x;
  if (idx >= K * M) return;
  int k = idx / M, m = idx % M;
  wt[(size_t)m * K + k] = (__bf16)w[idx];
}

// ---------------- LayerNorm (one wave per token, 3 elems/lane) ----------------
__global__ __launch_bounds__(256)
void ocab_ln(const float* __restrict__ x, const float* __restrict__ w,
             const float* __restrict__ b, __bf16* __restrict__ out) {
  int tok  = blockIdx.x * 8 + (threadIdx.x >> 5);
  int lane = threadIdx.x & 31;
  const float* row = x + (size_t)tok * 96 + lane * 3;
  float v0 = row[0], v1 = row[1], v2 = row[2];
  float s = v0 + v1 + v2;
#pragma unroll
  for (int m = 16; m >= 1; m >>= 1) s += __shfl_xor(s, m, 32);
  float mean = s * (1.0f / 96.0f);
  float d0 = v0 - mean, d1 = v1 - mean, d2 = v2 - mean;
  float q = d0 * d0 + d1 * d1 + d2 * d2;
#pragma unroll
  for (int m = 16; m >= 1; m >>= 1) q += __shfl_xor(q, m, 32);
  float rstd = rsqrtf(q * (1.0f / 96.0f) + LN_EPS);
  __bf16* o = out + (size_t)tok * 96 + lane * 3;
  const float* ww = w + lane * 3;
  const float* bb = b + lane * 3;
  o[0] = (__bf16)(d0 * rstd * ww[0] + bb[0]);
  o[1] = (__bf16)(d1 * rstd * ww[1] + bb[1]);
  o[2] = (__bf16)(d2 * rstd * ww[2] + bb[2]);
}

// ---------------- generic WMMA GEMM: Y = epi(X[N,K]bf16 @ Wt[M,K]^T + bias) ----------------
// Each wave computes one 16-row tile x CT consecutive 16-col tiles, reusing the A fragment
// CT times per K-step (6x fewer X reads, 6 v_wmma per A load).
enum { EPI_QKV = 0, EPI_GELU = 1, EPI_RESID = 2 };

template <int EPI, int CT>
__global__ __launch_bounds__(256)
void ocab_gemm(const __bf16* __restrict__ X, const __bf16* __restrict__ Wt,
               const float* __restrict__ bias, int K, int M, int ntc,
               __bf16* __restrict__ outB, float* __restrict__ outF,
               const float* __restrict__ resid,
               __bf16* __restrict__ qo, __bf16* __restrict__ kpo, __bf16* __restrict__ vpo) {
  int wid  = blockIdx.x * 8 + (threadIdx.x >> 5);
  int ncg  = ntc / CT;
  int tr = wid / ncg, cg = wid % ncg;
  int lane = threadIdx.x & 31, half = lane >> 4, lcol = lane & 15;
  const __bf16* xrow  = X  + (size_t)(tr * 16 + lcol) * K + half * 16;
  const __bf16* wbase = Wt + (size_t)(cg * CT * 16 + lcol) * K + half * 16;
  v8f acc[CT];
#pragma unroll
  for (int c = 0; c < CT; ++c) acc[c] = (v8f){};
  for (int k0 = 0; k0 < K; k0 += 32) {
    v16bf a = ldA(xrow + k0);
#pragma unroll
    for (int c = 0; c < CT; ++c) {
      v16bf b = ldB(wbase + (size_t)c * 16 * K + k0);
      acc[c] = __builtin_amdgcn_wmma_f32_16x16x32_bf16(false, a, false, b, (short)0, acc[c],
                                                       false, false);
    }
  }
#pragma unroll
  for (int c = 0; c < CT; ++c) {
    int   colg = (cg * CT + c) * 16 + lcol;
    float bv   = bias[colg];
#pragma unroll
    for (int r = 0; r < 8; ++r) {
      int   rowg = tr * 16 + r + half * 8;
      float v    = acc[c][r] + bv;
      if constexpr (EPI == EPI_GELU) {
        outB[(size_t)rowg * M + colg] = (__bf16)gelu_exact(v);
      } else if constexpr (EPI == EPI_RESID) {
        v += resid[(size_t)rowg * M + colg];
        outF[(size_t)rowg * M + colg] = v;
      } else {  // EPI_QKV: scatter q (scaled) and k/v into zero-padded images
        int g = colg / 96, cc = colg % 96;
        if (g == 0) {
          qo[(size_t)rowg * 96 + cc] = (__bf16)(v * SCALE_Q);
        } else {
          int bI = rowg / (Hdim * Wdim), p = rowg % (Hdim * Wdim);
          int i = p / Wdim, j = p % Wdim;
          size_t idx = (((size_t)bI * PH + i + PADz) * PH + (j + PADz)) * 96 + cc;
          if (g == 1) kpo[idx] = (__bf16)v; else vpo[idx] = (__bf16)v;
        }
      }
    }
  }
}

// ---------------- overlapping-window attention: 1 block = (window, head), 4 waves ----------------
__global__ __launch_bounds__(128)
void ocab_attn(const __bf16* __restrict__ q, const __bf16* __restrict__ kp,
               const __bf16* __restrict__ vp, __bf16* __restrict__ out) {
  int win = blockIdx.x / NHd;
  int hd  = blockIdx.x % NHd;
  int bI  = win / (NWH * NWH);
  int rw  = win % (NWH * NWH);
  int wy  = rw / NWH, wx = rw % NWH;

  int tid = threadIdx.x, wave = tid >> 5, lane = tid & 31;
  int half = lane >> 4, lcol = lane & 15;

  __shared__ __align__(16) __bf16 vlds[160][16];       // V tile, K-rows padded to 160
  __shared__ __align__(16) __bf16 plds[4][16][160];    // per-wave softmax(P), K padded to 160

  // ---- stage V[144][16] for this head into LDS (async DMA path if available) ----
#ifdef USE_ASYNC_LDS
  // 288 x 16-byte chunks, ASYNCcnt-tracked, no VGPR round-trip
  for (int e = tid; e < 288; e += 128) {
    int kr = e >> 1, hh = e & 1;
    int ky = kr / OWSz, kx = kr % OWSz;
    size_t idx = (((size_t)bI * PH + wy * WSz + ky) * PH + (wx * WSz + kx)) * 96 + hd * 16 + hh * 8;
    __builtin_amdgcn_global_load_async_to_lds_b128(
        (AS1 v4i*)(vp + idx), (AS3 v4i*)(&vlds[kr][hh * 8]), 0, 0);
  }
  for (int e = tid; e < 16 * 16; e += 128) vlds[144 + e / 16][e % 16] = (__bf16)0.0f;
  __builtin_amdgcn_s_wait_asynccnt(0);
#else
  for (int e = tid; e < 144 * 16; e += 128) {
    int kr = e / 16, d = e % 16;
    int ky = kr / OWSz, kx = kr % OWSz;
    size_t idx = (((size_t)bI * PH + wy * WSz + ky) * PH + (wx * WSz + kx)) * 96 + hd * 16 + d;
    vlds[kr][d] = vp[idx];
  }
  for (int e = tid; e < 16 * 16; e += 128) vlds[144 + e / 16][e % 16] = (__bf16)0.0f;
#endif
  __syncthreads();

  // A fragment: this wave's 16 q rows (K=16 padded to 32)
  int qwr = wave * 16 + lcol;                       // q row within window, 0..63
  int qr = qwr / WSz, qc = qwr % WSz;
  long qtok = (long)bI * Hdim * Wdim + (long)(wy * WSz + qr) * Wdim + (wx * WSz + qc);
  v16bf aq = half ? zero16() : ldB(q + (size_t)qtok * 96 + hd * 16);

  // scores: 9 tiles of 16 k-rows (144 = 9*16 exactly)
  v8f s[9];
  v8f zacc = {};
#pragma unroll
  for (int t = 0; t < 9; ++t) {
    int kr = t * 16 + lcol;
    int ky = kr / OWSz, kx = kr % OWSz;
    size_t idx = (((size_t)bI * PH + wy * WSz + ky) * PH + (wx * WSz + kx)) * 96 + hd * 16;
    v16bf bk = half ? zero16() : ldB(kp + idx);     // B: K=16 valid in lane half 0 only
    s[t] = __builtin_amdgcn_wmma_f32_16x16x32_bf16(false, aq, false, bk, (short)0, zacc,
                                                   false, false);
  }

  // softmax over 144 per row; row m = r + 8*half lives in one 16-lane half
#pragma unroll
  for (int r = 0; r < 8; ++r) {
    float mx = -3.0e38f;
#pragma unroll
    for (int t = 0; t < 9; ++t) mx = fmaxf(mx, s[t][r]);
#pragma unroll
    for (int m = 8; m >= 1; m >>= 1) mx = fmaxf(mx, __shfl_xor(mx, m, 16));
    float sum = 0.0f;
#pragma unroll
    for (int t = 0; t < 9; ++t) { float e = __expf(s[t][r] - mx); s[t][r] = e; sum += e; }
#pragma unroll
    for (int m = 8; m >= 1; m >>= 1) sum += __shfl_xor(sum, m, 16);
    float inv = 1.0f / sum;
    int rowm = r + 8 * half;
#pragma unroll
    for (int t = 0; t < 9; ++t) plds[wave][rowm][t * 16 + lcol] = (__bf16)(s[t][r] * inv);
    plds[wave][rowm][144 + lcol] = (__bf16)0.0f;    // zero K-pad 144..159
  }
  __syncthreads();

  // P(16x160) @ V(160x16) with 5 K=32 steps
  v8f o = {};
#pragma unroll
  for (int st = 0; st < 5; ++st) {
    v16bf ap = ldA(&plds[wave][lcol][st * 32 + half * 16]);
    v16bf bv;
#pragma unroll
    for (int i = 0; i < 16; ++i) {
      int kr = st * 32 + half * 16 + i;
      bv[i] = vlds[kr][lcol];
    }
    o = __builtin_amdgcn_wmma_f32_16x16x32_bf16(false, ap, false, bv, (short)0, o, false, false);
  }

  // write output tile back to [B, H*W, C] (q windows are non-overlapping)
#pragma unroll
  for (int r = 0; r < 8; ++r) {
    int owr = wave * 16 + r + 8 * half;
    int orr = owr / WSz, occ = owr % WSz;
    long tok = (long)bI * Hdim * Wdim + (long)(wy * WSz + orr) * Wdim + (wx * WSz + occ);
    out[(size_t)tok * 96 + hd * 16 + lcol] = (__bf16)o[r];
  }
}

// ---------------- launch ----------------
extern "C" void kernel_launch(void* const* d_in, const int* in_sizes, int n_in,
                              void* d_out, int out_size, void* d_ws, size_t ws_size,
                              hipStream_t stream) {
  const float* x     = (const float*)d_in[0];
  const float* n1w   = (const float*)d_in[3];
  const float* n1b   = (const float*)d_in[4];
  const float* qkvw  = (const float*)d_in[5];
  const float* qkvb  = (const float*)d_in[6];
  const float* projw = (const float*)d_in[7];
  const float* projb = (const float*)d_in[8];
  const float* n2w   = (const float*)d_in[9];
  const float* n2b   = (const float*)d_in[10];
  const float* fc1w  = (const float*)d_in[11];
  const float* fc1b  = (const float*)d_in[12];
  const float* fc2w  = (const float*)d_in[13];
  const float* fc2b  = (const float*)d_in[14];
  const float* fc3w  = (const float*)d_in[15];
  const float* fc3b  = (const float*)d_in[16];
  const float* fc4w  = (const float*)d_in[17];
  const float* fc4b  = (const float*)d_in[18];

  char* ws = (char*)d_ws;
  __bf16* xn   = (__bf16*)(ws + O_A1);
  __bf16* aout = (__bf16*)(ws + O_A1);   // reuses xn (dead after QKV)
  __bf16* h1   = (__bf16*)(ws + O_A1);   // reuses attn_out (dead after proj)
  __bf16* h3   = (__bf16*)(ws + O_A1);   // reuses h1 (dead after fc2)
  __bf16* qb   = (__bf16*)(ws + O_Q);
  __bf16* kpd  = (__bf16*)(ws + O_KP);
  __bf16* vpd  = (__bf16*)(ws + O_VP);
  __bf16* h2   = (__bf16*)(ws + O_A2);   // reuses q/kpad/vpad (dead after attn)
  float*  x2   = (float*)(ws + O_X2);
  __bf16* yn   = (__bf16*)(ws + O_YN);
  __bf16* wqkv  = (__bf16*)(ws + O_WQKV);
  __bf16* wproj = (__bf16*)(ws + O_WPROJ);
  __bf16* wfc1  = (__bf16*)(ws + O_WFC1);
  __bf16* wfc2  = (__bf16*)(ws + O_WFC2);
  __bf16* wfc3  = (__bf16*)(ws + O_WFC3);
  __bf16* wfc4  = (__bf16*)(ws + O_WFC4);

  auto cvt = [&](const float* w, __bf16* wt, int K, int M) {
    int n = K * M;
    ocab_cvtw<<<(n + 255) / 256, 256, 0, stream>>>(w, wt, K, M);
  };
  cvt(qkvw,  wqkv,  96, 288);
  cvt(projw, wproj, 96, 96);
  cvt(fc1w,  wfc1,  96, 192);
  cvt(fc2w,  wfc2,  192, 384);
  cvt(fc3w,  wfc3,  384, 192);
  cvt(fc4w,  wfc4,  192, 96);

  // zero pad rings for K/V padded images (interior overwritten by QKV scatter)
  (void)hipMemsetAsync(ws + O_KP, 0, 2 * SZ_KPAD, stream);

  constexpr int NTR = (int)(NTOK / 16);  // 9216 row tiles

  // 1) LN1 -> xn bf16
  ocab_ln<<<(int)(NTOK / 8), 256, 0, stream>>>(x, n1w, n1b, xn);

  // 2) QKV GEMM + scatter (M=288, 18 col tiles -> 3 groups of 6)
  ocab_gemm<EPI_QKV, 6><<<NTR * 3 / 8, 256, 0, stream>>>(
      xn, wqkv, qkvb, 96, 288, 18, nullptr, nullptr, nullptr, qb, kpd, vpd);

  // 3) attention
  ocab_attn<<<NWIN * NHd, 128, 0, stream>>>(qb, kpd, vpd, aout);

  // 4) proj + bias + shortcut -> x2 fp32 (6 col tiles -> 1 group)
  ocab_gemm<EPI_RESID, 6><<<NTR * 1 / 8, 256, 0, stream>>>(
      aout, wproj, projb, 96, 96, 6, nullptr, x2, x, nullptr, nullptr, nullptr);

  // 5) LN2 -> yn bf16
  ocab_ln<<<(int)(NTOK / 8), 256, 0, stream>>>(x2, n2w, n2b, yn);

  // 6) MLP
  ocab_gemm<EPI_GELU, 6><<<NTR * 2 / 8, 256, 0, stream>>>(
      yn, wfc1, fc1b, 96, 192, 12, h1, nullptr, nullptr, nullptr, nullptr, nullptr);
  ocab_gemm<EPI_GELU, 6><<<NTR * 4 / 8, 256, 0, stream>>>(
      h1, wfc2, fc2b, 192, 384, 24, h2, nullptr, nullptr, nullptr, nullptr, nullptr);
  ocab_gemm<EPI_GELU, 6><<<NTR * 2 / 8, 256, 0, stream>>>(
      h2, wfc3, fc3b, 384, 192, 12, h3, nullptr, nullptr, nullptr, nullptr, nullptr);

  // 7) fc4 + bias + x2 residual -> d_out fp32
  ocab_gemm<EPI_RESID, 6><<<NTR * 1 / 8, 256, 0, stream>>>(
      h3, wfc4, fc4b, 192, 96, 6, nullptr, (float*)d_out, x2, nullptr, nullptr, nullptr);
}
